// MoELayer_38912403702357
// MI455X (gfx1250) — matrix operand order repaired and checked by reading, not compiled, not run on hardware
//
#include <hip/hip_runtime.h>
#include <hip/hip_bf16.h>
#include <stdint.h>

// ---------------------------------------------------------------------------
// MoE layer for MI455X (gfx1250, wave32, WMMA + TDM).
// B=4 S=2048 D=1024 -> T=8192 tokens. E=8 experts, top-2, cap=2560, H=1365.
// bf16 WMMA for all GEMMs; weights pre-padded to Hp=1408 so inner loops are
// branch-free; A panels staged in LDS via tensor_load_to_lds gather mode
// (TDM is the natural MoE token-gather primitive), 2-phase K staging.
// ---------------------------------------------------------------------------

typedef __bf16 bf16;
typedef __attribute__((ext_vector_type(16))) __bf16 v16bf;
typedef __attribute__((ext_vector_type(8)))  __bf16 v8bf;
typedef __attribute__((ext_vector_type(8)))  float  v8f;
typedef __attribute__((ext_vector_type(4)))  unsigned int v4u;
typedef __attribute__((ext_vector_type(8)))  int v8i;
typedef __attribute__((ext_vector_type(4)))  int v4i;

#define T_TOK   8192
#define D_DIM   1024
#define E_EXP   8
#define H_DIM   1365
#define HP_DIM  1408      // H padded to multiple of 64 (22 N-tiles of 64)
#define CAP     2560      // int(T*TOPK/E*CF) = 8192*2/8*1.25
#define KPH     512       // K per staging phase (32 rows x 512 = 32 KB LDS)

#if defined(__has_builtin)
#if __has_builtin(__builtin_amdgcn_tensor_load_to_lds) && \
    __has_builtin(__builtin_amdgcn_s_wait_tensorcnt)
#define HAVE_TDM 1
#endif
#endif

// -------------------------------------------------------------- conversions
__global__ __launch_bounds__(256) void moe_cvt_bf16(const float* __restrict__ in,
                                                    bf16* __restrict__ out, long n) {
  long i = (long)blockIdx.x * blockDim.x + threadIdx.x;
  long stride = (long)gridDim.x * blockDim.x;
  for (; i < n; i += stride) out[i] = (bf16)in[i];
}

// [E][H][D] f32  ->  [E][Hp][D] bf16, zero rows h >= H
__global__ __launch_bounds__(256)
void moe_cvt_pad_rows(const float* __restrict__ in, bf16* __restrict__ out,
                      int experts) {
  const long per = (long)HP_DIM * D_DIM;
  const long n = experts * per;
  long i = (long)blockIdx.x * blockDim.x + threadIdx.x;
  long stride = (long)gridDim.x * blockDim.x;
  for (; i < n; i += stride) {
    long e = i / per, rem = i % per;
    long h = rem / D_DIM, d = rem % D_DIM;
    out[i] = (h < H_DIM) ? (bf16)in[(e * H_DIM + h) * D_DIM + d] : (bf16)0.0f;
  }
}

// [E][D][H] f32  ->  [E][D][Hp] bf16, zero cols k >= H
__global__ __launch_bounds__(256)
void moe_cvt_pad_cols(const float* __restrict__ in, bf16* __restrict__ out,
                      int experts) {
  const long per = (long)D_DIM * HP_DIM;
  const long n = experts * per;
  long i = (long)blockIdx.x * blockDim.x + threadIdx.x;
  long stride = (long)gridDim.x * blockDim.x;
  for (; i < n; i += stride) {
    long e = i / per, rem = i % per;
    long d = rem / HP_DIM, k = rem % HP_DIM;
    out[i] = (k < H_DIM) ? (bf16)in[(e * D_DIM + d) * H_DIM + k] : (bf16)0.0f;
  }
}

__global__ void moe_zero_stats(float* __restrict__ p) {
  if (threadIdx.x < 16) p[threadIdx.x] = 0.0f;   // [0..7]=sum(probs) [8..15]=counts
}

// ------------------------------------------------------------------- router
__global__ __launch_bounds__(256)
void moe_router(const float* __restrict__ x, const float* __restrict__ rw,
                int* __restrict__ tok_top, float* __restrict__ tok_gate,
                float* __restrict__ stats) {
  const int wave = threadIdx.x >> 5;
  const int lane = threadIdx.x & 31;
  const int t = blockIdx.x * 8 + wave;
  if (t >= T_TOK) return;

  float acc[E_EXP];
#pragma unroll
  for (int e = 0; e < E_EXP; ++e) acc[e] = 0.0f;
  const float* xr = x + (long)t * D_DIM;
  for (int d = lane; d < D_DIM; d += 32) {
    float xv = xr[d];
#pragma unroll
    for (int e = 0; e < E_EXP; ++e) acc[e] += xv * rw[e * D_DIM + d];
  }
#pragma unroll
  for (int e = 0; e < E_EXP; ++e) {
#pragma unroll
    for (int off = 16; off > 0; off >>= 1) acc[e] += __shfl_xor(acc[e], off, 32);
  }
  if (lane == 0) {
    float m = acc[0];
#pragma unroll
    for (int e = 1; e < E_EXP; ++e) m = fmaxf(m, acc[e]);
    float p[E_EXP], s = 0.0f;
#pragma unroll
    for (int e = 0; e < E_EXP; ++e) { p[e] = __expf(acc[e] - m); s += p[e]; }
    float inv = 1.0f / s;
#pragma unroll
    for (int e = 0; e < E_EXP; ++e) p[e] *= inv;
    int i0 = 0;
#pragma unroll
    for (int e = 1; e < E_EXP; ++e) if (p[e] > p[i0]) i0 = e;
    int i1 = (i0 == 0) ? 1 : 0;
#pragma unroll
    for (int e = 0; e < E_EXP; ++e) if (e != i0 && p[e] > p[i1]) i1 = e;
    float wsum = p[i0] + p[i1];
    tok_top[2 * t] = i0;  tok_top[2 * t + 1] = i1;
    tok_gate[2 * t] = p[i0] / wsum;  tok_gate[2 * t + 1] = p[i1] / wsum;
#pragma unroll
    for (int e = 0; e < E_EXP; ++e) atomicAdd(&stats[e], p[e]);
    atomicAdd(&stats[8 + i0], 1.0f);
    atomicAdd(&stats[8 + i1], 1.0f);
  }
}

// ------------------------------------------------- capacity scan + compact
__global__ __launch_bounds__(256)
void moe_capacity(const int* __restrict__ tok_top, const float* __restrict__ tok_gate,
                  int* __restrict__ list_tok, float* __restrict__ list_gate,
                  int* __restrict__ cnt_capped) {
  __shared__ int scan[256];
  const int e = blockIdx.x;
  const int tid = threadIdx.x;
  int running = 0;
  for (int base = 0; base < T_TOK; base += 256) {
    int t = base + tid;
    int flag = 0; float g = 0.0f;
    if (tok_top[2 * t] == e)          { flag = 1; g = tok_gate[2 * t]; }
    else if (tok_top[2 * t + 1] == e) { flag = 1; g = tok_gate[2 * t + 1]; }
    scan[tid] = flag;
    __syncthreads();
    for (int off = 1; off < 256; off <<= 1) {
      int v = (tid >= off) ? scan[tid - off] : 0;
      __syncthreads();
      scan[tid] += v;
      __syncthreads();
    }
    int pos = running + scan[tid];
    if (flag && pos <= CAP) {
      list_tok [e * CAP + pos - 1] = t;
      list_gate[e * CAP + pos - 1] = g;
    }
    running += scan[255];
    __syncthreads();
  }
  int cnt = running < CAP ? running : CAP;
  if (tid == 0) cnt_capped[e] = cnt;
  for (int i = cnt + tid; i < CAP; i += 256) {
    list_tok [e * CAP + i] = 0;
    list_gate[e * CAP + i] = 0.0f;
  }
}

// -------------------------------------------------- TDM gather: 16 rows->LDS
// Gather-mode Tensor DMA descriptor (CDNA5 ISA):
// 16-bit row indices, data_size=2B, row width KPH elems, row stride D elems.
// NOTE: this toolchain exposes the 6-arg builtin
//   (uint32x4 g0, int32x8 g1, int32x4 g2, int32x4 g3, int32x8, i32 cpol)
__device__ __forceinline__ void tdm_gather16(const bf16* gsrc, unsigned lds_byte_off,
                                             const int* idx) {
#ifdef HAVE_TDM
  unsigned long long ga = (unsigned long long)(uintptr_t)gsrc;
  v4u g0 = { 0x80000001u,                     // count=1 | gather_mode | 16b idx
             lds_byte_off,
             (unsigned)(ga & 0xFFFFFFFFu),
             (unsigned)((ga >> 32) & 0x01FFFFFFu) | 0x80000000u };  // type=2
  v8i g1 = { 0x00010000,                      // data_size = 2 bytes
             (int)((unsigned)D_DIM << 16),    // tensor_dim0 lo16 @ bits 63:48
             (int)((unsigned)T_TOK << 16),    // tensor_dim1 lo16 @ bits 95:80
             (int)((unsigned)KPH  << 16),     // tile_dim0 @ bits 127:112
             16,                              // tile_dim1 = #valid indices
             D_DIM,                           // tensor_dim0_stride lo32
             0, 0 };
  v4i g2 = { (idx[1]  << 16) | (idx[0]  & 0xFFFF),
             (idx[3]  << 16) | (idx[2]  & 0xFFFF),
             (idx[5]  << 16) | (idx[4]  & 0xFFFF),
             (idx[7]  << 16) | (idx[6]  & 0xFFFF) };
  v4i g3 = { (idx[9]  << 16) | (idx[8]  & 0xFFFF),
             (idx[11] << 16) | (idx[10] & 0xFFFF),
             (idx[13] << 16) | (idx[12] & 0xFFFF),
             (idx[15] << 16) | (idx[14] & 0xFFFF) };
  v8i gx = { 0, 0, 0, 0, 0, 0, 0, 0 };        // unused trailing group
  __builtin_amdgcn_tensor_load_to_lds(g0, g1, g2, g3, gx, 0);
#else
  (void)gsrc; (void)lds_byte_off; (void)idx;
#endif
}

// -------------------------------------------------------------- GEMM 1 (up)
// h[r,c] = silu(A.w1[c,:]) * (A.w3[c,:]); A rows gathered by TDM into LDS.
// Block = 8 waves: 32(M) x 64(N) tile, K staged in 2 phases of 512.
__global__ __launch_bounds__(256)
void moe_gemm1_swiglu(const bf16* __restrict__ xb, const int* __restrict__ toklist,
                      const bf16* __restrict__ w1p, const bf16* __restrict__ w3p,
                      bf16* __restrict__ hout) {
  __shared__ bf16 ldsA[32 * KPH];            // 32 KB
  const int lane = threadIdx.x & 31;
  const int wave = threadIdx.x >> 5;
  const int half = lane >> 4;
  const int l16  = lane & 15;
  const int wm = wave & 1, wn = wave >> 1;
  const int rowBase = blockIdx.x * 32 + wm * 16;
  const int colBase = blockIdx.y * 64 + wn * 16;

  const int  myCol = colBase + l16;          // padded weights: always valid
  const bf16* b1row = w1p + (long)myCol * D_DIM;
  const bf16* b3row = w3p + (long)myCol * D_DIM;
  const int ldsRow = wm * 16 + l16;
  const unsigned ldsBase = (unsigned)(uintptr_t)(void*)ldsA;

  v8f c1 = {}; v8f c3 = {};
  for (int phase = 0; phase < 2; ++phase) {
    const int kbase = phase * KPH;
#ifdef HAVE_TDM
    if (wave == 0) {
      for (int d = 0; d < 2; ++d) {
        const int rs = blockIdx.x * 32 + d * 16;
        int idx[16];
#pragma unroll
        for (int i = 0; i < 16; ++i)
          idx[i] = toklist ? toklist[rs + i] : (rs + i);
        tdm_gather16(xb + kbase, ldsBase + (unsigned)(d * 16 * KPH * 2), idx);
      }
      __builtin_amdgcn_s_wait_tensorcnt(0);
    }
#else
    // cooperative fallback: 256 threads copy 32 rows x KPH bf16
    for (int i = threadIdx.x; i < 32 * KPH / 8; i += 256) {
      const int row = i >> 6, seg = i & 63;          // KPH/8 = 64 segments
      const long tok = toklist ? (long)toklist[blockIdx.x * 32 + row]
                               : (long)(blockIdx.x * 32 + row);
      *(v8bf*)&ldsA[row * KPH + seg * 8] =
          *(const v8bf*)(xb + tok * D_DIM + kbase + seg * 8);
    }
#endif
    __syncthreads();

    const bf16* ar = &ldsA[ldsRow * KPH];
    for (int k0 = 0; k0 < KPH; k0 += 32) {
      __builtin_prefetch(b1row + kbase + k0 + 128, 0, 3);
      v16bf a = *(const v16bf*)(ar + k0 + 16 * half);
      v16bf b1 = *(const v16bf*)(b1row + kbase + k0 + 16 * half);
      v16bf b3 = *(const v16bf*)(b3row + kbase + k0 + 16 * half);
      c1 = __builtin_amdgcn_wmma_f32_16x16x32_bf16(false, a, false, b1,
                                                   (short)0, c1, false, false);
      c3 = __builtin_amdgcn_wmma_f32_16x16x32_bf16(false, a, false, b3,
                                                   (short)0, c3, false, false);
    }
    __syncthreads();
  }
#pragma unroll
  for (int r = 0; r < 8; ++r) {
    const int row = rowBase + r + 8 * half;
    float z1 = c1[r], z3 = c3[r];
    float s = z1 / (1.0f + __expf(-z1));
    hout[(long)row * HP_DIM + myCol] = (bf16)(s * z3);
  }
}

// ------------------------------------------------------------ GEMM 2 (down)
// y[r,d] = h[r,:].w2p[d,:]; wave tile 16(M) x 32(N): one A, two B, two acc.
__global__ __launch_bounds__(256)
void moe_gemm2_down(const bf16* __restrict__ hbuf, const bf16* __restrict__ w2p,
                    const int* __restrict__ toklist, const float* __restrict__ gatelist,
                    float* __restrict__ outp, int accumulate) {
  const int lane = threadIdx.x & 31;
  const int wave = threadIdx.x >> 5;
  const int half = lane >> 4;
  const int l16  = lane & 15;
  const int wm = wave & 1, wn = wave >> 1;
  const int rowBase = blockIdx.x * 32 + wm * 16;
  const int colBase = blockIdx.y * 128 + wn * 32;

  const bf16* arow = hbuf + (long)(rowBase + l16) * HP_DIM;
  const bf16* w2r0 = w2p + (long)(colBase      + l16) * HP_DIM;
  const bf16* w2r1 = w2p + (long)(colBase + 16 + l16) * HP_DIM;

  v8f c0 = {}; v8f c1 = {};
  for (int k0 = 0; k0 < HP_DIM; k0 += 32) {
    __builtin_prefetch(arow + k0 + 128, 0, 3);
    v16bf a = *(const v16bf*)(arow + k0 + 16 * half);
    v16bf b0 = *(const v16bf*)(w2r0 + k0 + 16 * half);
    v16bf b1 = *(const v16bf*)(w2r1 + k0 + 16 * half);
    c0 = __builtin_amdgcn_wmma_f32_16x16x32_bf16(false, a, false, b0,
                                                 (short)0, c0, false, false);
    c1 = __builtin_amdgcn_wmma_f32_16x16x32_bf16(false, a, false, b1,
                                                 (short)0, c1, false, false);
  }
#pragma unroll
  for (int r = 0; r < 8; ++r) {
    const int  row = rowBase + r + 8 * half;
    const long tk  = toklist ? (long)toklist[row] : (long)row;
    const float g  = gatelist ? gatelist[row] : 1.0f;
    float* d0 = outp + tk * D_DIM + colBase + l16;
    float* d1 = d0 + 16;
    if (accumulate) { *d0 += g * c0[r]; *d1 += g * c1[r]; }
    else            { *d0  = c0[r];     *d1  = c1[r];     }
  }
}

// ---------------------------------------------------------------- finalize
__global__ void moe_finalize(const float* __restrict__ stats, const int* __restrict__ cnt,
                             float* __restrict__ outp) {
  if (threadIdx.x == 0 && blockIdx.x == 0) {
    (void)cnt;
    float aux = 0.0f, used = 0.0f;
    const float Tf = (float)T_TOK;
#pragma unroll
    for (int e = 0; e < E_EXP; ++e) {
      aux  += (stats[e] / Tf) * (stats[8 + e] / Tf);
      used += (stats[8 + e] > 0.0f) ? 1.0f : 0.0f;
    }
    outp[(long)T_TOK * D_DIM]     = (float)E_EXP * aux;
    outp[(long)T_TOK * D_DIM + 1] = 100.0f * used / (float)E_EXP;
  }
}

// ---------------------------------------------------------------- launcher
static inline void* carve(char*& p, size_t bytes) {
  void* r = (void*)p;
  p += (bytes + 255) & ~(size_t)255;
  return r;
}

extern "C" void kernel_launch(void* const* d_in, const int* in_sizes, int n_in,
                              void* d_out, int out_size, void* d_ws, size_t ws_size,
                              hipStream_t stream) {
  (void)in_sizes; (void)n_in; (void)out_size; (void)ws_size;
  const float* x   = (const float*)d_in[0];
  const float* rw  = (const float*)d_in[1];
  const float* w1  = (const float*)d_in[2];
  const float* w2  = (const float*)d_in[3];
  const float* w3  = (const float*)d_in[4];
  const float* sw1 = (const float*)d_in[5];
  const float* sw2 = (const float*)d_in[6];
  const float* sw3 = (const float*)d_in[7];
  float* outp = (float*)d_out;

  char* ws = (char*)d_ws;
  const long nX   = (long)T_TOK * D_DIM;
  const long nWP  = (long)E_EXP * HP_DIM * D_DIM;   // padded expert weights
  const long nSWP = (long)HP_DIM * D_DIM;

  bf16* xb   = (bf16*)carve(ws, nX * sizeof(bf16));
  bf16* w1p  = (bf16*)carve(ws, nWP * sizeof(bf16));
  bf16* w2p  = (bf16*)carve(ws, nWP * sizeof(bf16));
  bf16* w3p  = (bf16*)carve(ws, nWP * sizeof(bf16));
  bf16* sw1p = (bf16*)carve(ws, nSWP * sizeof(bf16));
  bf16* sw2p = (bf16*)carve(ws, nSWP * sizeof(bf16));
  bf16* sw3p = (bf16*)carve(ws, nSWP * sizeof(bf16));
  int*   tok_top    = (int*)  carve(ws, (size_t)T_TOK * 2 * sizeof(int));
  float* tok_gate   = (float*)carve(ws, (size_t)T_TOK * 2 * sizeof(float));
  int*   list_tok   = (int*)  carve(ws, (size_t)E_EXP * CAP * sizeof(int));
  float* list_gate  = (float*)carve(ws, (size_t)E_EXP * CAP * sizeof(float));
  int*   cnt_capped = (int*)  carve(ws, (size_t)E_EXP * sizeof(int));
  float* stats      = (float*)carve(ws, 16 * sizeof(float));
  bf16*  hbuf       = (bf16*) carve(ws, (size_t)CAP * HP_DIM * sizeof(bf16));
  bf16*  hshared    = (bf16*) carve(ws, (size_t)T_TOK * HP_DIM * sizeof(bf16));

  moe_zero_stats<<<1, 32, 0, stream>>>(stats);

  // f32 -> bf16 (+ padding) conversions
  moe_cvt_bf16<<<2048, 256, 0, stream>>>(x, xb, nX);
  moe_cvt_pad_rows<<<4096, 256, 0, stream>>>(w1, w1p, E_EXP);
  moe_cvt_pad_rows<<<4096, 256, 0, stream>>>(w3, w3p, E_EXP);
  moe_cvt_pad_cols<<<4096, 256, 0, stream>>>(w2, w2p, E_EXP);
  moe_cvt_pad_rows<<<1024, 256, 0, stream>>>(sw1, sw1p, 1);
  moe_cvt_pad_rows<<<1024, 256, 0, stream>>>(sw3, sw3p, 1);
  moe_cvt_pad_cols<<<1024, 256, 0, stream>>>(sw2, sw2p, 1);

  // router + capacity compaction
  moe_router<<<T_TOK / 8, 256, 0, stream>>>(x, rw, tok_top, tok_gate, stats);
  moe_capacity<<<E_EXP, 256, 0, stream>>>(tok_top, tok_gate, list_tok, list_gate,
                                          cnt_capped);

  // shared expert (dense): writes final; experts accumulate on top
  moe_gemm1_swiglu<<<dim3(T_TOK / 32, HP_DIM / 64), 256, 0, stream>>>(
      xb, nullptr, sw1p, sw3p, hshared);
  moe_gemm2_down<<<dim3(T_TOK / 32, D_DIM / 128), 256, 0, stream>>>(
      hshared, sw2p, nullptr, nullptr, outp, /*accumulate=*/0);

  // routed experts (stream-serialized, hbuf reused)
  for (int e = 0; e < E_EXP; ++e) {
    const bf16* e1 = w1p + (long)e * HP_DIM * D_DIM;
    const bf16* e3 = w3p + (long)e * HP_DIM * D_DIM;
    const bf16* e2 = w2p + (long)e * D_DIM * HP_DIM;
    const int*   lt = list_tok  + (long)e * CAP;
    const float* lg = list_gate + (long)e * CAP;
    moe_gemm1_swiglu<<<dim3(CAP / 32, HP_DIM / 64), 256, 0, stream>>>(
        xb, lt, e1, e3, hbuf);
    moe_gemm2_down<<<dim3(CAP / 32, D_DIM / 128), 256, 0, stream>>>(
        hbuf, e2, lt, lg, outp, /*accumulate=*/1);
  }

  moe_finalize<<<1, 32, 0, stream>>>(stats, cnt_capped, outp);
}
